// GPSModel_45964740002209
// MI455X (gfx1250) — matrix operand order repaired and checked by reading, not compiled, not run on hardware
//
#include <hip/hip_runtime.h>

// ---------------- types / helpers ----------------
typedef __attribute__((ext_vector_type(16))) __bf16        bf16x16;
typedef __attribute__((ext_vector_type(8)))  float         floatx8;
typedef __attribute__((ext_vector_type(4)))  unsigned int  u32x4;

union BFrag { bf16x16 v; u32x4 q[2]; };

static __device__ __forceinline__ unsigned short f2bf(float f) {
  unsigned int u = __float_as_uint(f);
  u += 0x7FFFu + ((u >> 16) & 1u);              // round-to-nearest-even
  return (unsigned short)(u >> 16);
}

// ---------------- problem constants ----------------
constexpr int H      = 256;     // hidden
constexpr int BOND_D = 320;
constexpr int NA     = 16385;   // atoms incl. pad row 0
constexpr int NB     = 65537;   // bonds incl. pad row 0
constexpr int MP_A   = 16400;   // NA padded to /16
constexpr int MP_B   = 65552;   // NB padded to /16
constexpr int NMOL   = 256;
constexpr int ASZ    = 64;
constexpr int MAXNB  = 6;

// ---------------- elementwise kernels ----------------
__global__ void k_conv_flat(const float* __restrict__ s, unsigned short* __restrict__ d, long n) {
  long i = (long)blockIdx.x * blockDim.x + threadIdx.x;
  if (i < n) d[i] = f2bf(s[i]);
}

__global__ void k_conv_pad(const float* __restrict__ s, unsigned short* __restrict__ d,
                           int rows_src, int rows_dst, int cols) {
  long i = (long)blockIdx.x * blockDim.x + threadIdx.x;
  long tot = (long)rows_dst * cols;
  if (i >= tot) return;
  int r = (int)(i / cols);
  float v = (r < rows_src) ? s[i] : 0.0f;
  d[i] = f2bf(v);
}

// ma_out = (ma_in? ma_in:0) + sum_j(mb[a2b[r][j]]) * max_j(mb[a2b[r][j]])   (elementwise per col)
__global__ void k_agg(const float* __restrict__ mb, const int* __restrict__ a2b,
                      const float* __restrict__ ma_in, float* __restrict__ out, int rows) {
  long i = (long)blockIdx.x * blockDim.x + threadIdx.x;
  long tot = (long)rows * H;
  if (i >= tot) return;
  int r = (int)(i / H), c = (int)(i % H);
  const int* nb = a2b + (long)r * MAXNB;
  float v0 = mb[(long)nb[0] * H + c];
  float s = v0, mx = v0;
#pragma unroll
  for (int j = 1; j < MAXNB; ++j) {
    float v = mb[(long)nb[j] * H + c];
    s += v; mx = fmaxf(mx, v);
  }
  float a = s * mx;
  out[i] = ma_in ? (ma_in[i] + a) : a;
}

// diff_bf[b] = bf16(ma[b2a[b]] - mb[b2revb[b]]) ; pad rows -> 0
__global__ void k_gatherdiff(const float* __restrict__ ma, const float* __restrict__ mb,
                             const int* __restrict__ b2a, const int* __restrict__ b2revb,
                             unsigned short* __restrict__ diff) {
  long i = (long)blockIdx.x * blockDim.x + threadIdx.x;
  long tot = (long)MP_B * H;
  if (i >= tot) return;
  int r = (int)(i / H), c = (int)(i % H);
  float v = 0.0f;
  if (r < NB) v = ma[(long)b2a[r] * H + c] - mb[(long)b2revb[r] * H + c];
  diff[i] = f2bf(v);
}

// cat_bf = [aggm | ma | ia] -> bf16, pad rows -> 0
__global__ void k_concat(const float* __restrict__ aggm, const float* __restrict__ ma,
                         const float* __restrict__ ia, unsigned short* __restrict__ cat) {
  long i = (long)blockIdx.x * blockDim.x + threadIdx.x;
  long tot = (long)MP_A * (3 * H);
  if (i >= tot) return;
  int r = (int)(i / (3 * H)), c = (int)(i % (3 * H));
  float v = 0.0f;
  if (r < NA) {
    if (c < H)           v = aggm[(long)r * H + c];
    else if (c < 2 * H)  v = ma[(long)r * H + (c - H)];
    else                 v = ia[(long)r * H + (c - 2 * H)];
  }
  cat[i] = f2bf(v);
}

// msg = relu(node + gru_bias); msg_bf holds rows 1..NA-1 (the GRU input x)
__global__ void k_msg(const float* __restrict__ node, const float* __restrict__ bias,
                      float* __restrict__ msg, unsigned short* __restrict__ msg_bf) {
  long i = (long)blockIdx.x * blockDim.x + threadIdx.x;
  long tot = (long)NA * H;
  if (i >= tot) return;
  int r = (int)(i / H), c = (int)(i % H);
  float v = fmaxf(node[i] + bias[c], 0.0f);
  msg[i] = v;
  if (r >= 1) msg_bf[(long)(r - 1) * H + c] = f2bf(v);
}

// h0[mol, c] = max_t node[1 + mol*ASZ + t, c]
__global__ void k_h0(const float* __restrict__ node, float* __restrict__ h0f,
                     unsigned short* __restrict__ h0b) {
  int i = blockIdx.x * blockDim.x + threadIdx.x;
  if (i >= NMOL * H) return;
  int mol = i / H, c = i % H;
  const float* p = node + (long)(1 + mol * ASZ) * H + c;
  float mx = p[0];
  for (int t = 1; t < ASZ; ++t) mx = fmaxf(mx, p[(long)t * H]);
  h0f[i] = mx;
  h0b[i] = f2bf(mx);
}

// gmsg row 0 = [msg[0] | msg[0]] ; pad rows NA..MP_A-1 -> 0
__global__ void k_gmsg_fixup(const float* __restrict__ msg, unsigned short* __restrict__ gmsg) {
  int i = blockIdx.x * blockDim.x + threadIdx.x;
  int tot = (1 + (MP_A - NA)) * (2 * H);
  if (i >= tot) return;
  int rr = i / (2 * H), c = i % (2 * H);
  if (rr == 0) gmsg[c] = f2bf(msg[c % H]);
  else         gmsg[(long)(NA + rr - 1) * (2 * H) + c] = 0;
}

// out = LN(f_atoms + ah) * gamma + beta   (one block per row, H threads)
__global__ __launch_bounds__(256) void
k_ln(const float* __restrict__ fa, const float* __restrict__ ah,
     const float* __restrict__ g, const float* __restrict__ b, float* __restrict__ out) {
  const int r = blockIdx.x, c = threadIdx.x;
  __shared__ float sm[H];
  const size_t idx = (size_t)r * H + c;
  const float x = fa[idx] + ah[idx];
  sm[c] = x; __syncthreads();
  for (int s = H / 2; s > 0; s >>= 1) { if (c < s) sm[c] += sm[c + s]; __syncthreads(); }
  const float mean = sm[0] * (1.0f / H);
  __syncthreads();
  const float d = x - mean;
  sm[c] = d * d; __syncthreads();
  for (int s = H / 2; s > 0; s >>= 1) { if (c < s) sm[c] += sm[c + s]; __syncthreads(); }
  const float var = sm[0] * (1.0f / H);
  out[idx] = d * rsqrtf(var + 1e-5f) * g[c] + b[c];
}

// ---------------- WMMA bf16 GEMM: C = act(A[M,K] @ W[N,K]^T + bias + addend) ----------------
// Wave computes a 16x64 tile (1 A-fragment reused over 4 B-fragments).
__global__ __launch_bounds__(256) void
k_gemm(const unsigned short* __restrict__ A, const unsigned short* __restrict__ W,
       const float* __restrict__ bias, const float* __restrict__ addend,
       float* __restrict__ C, int Mp, int Mv, int N, int K, int do_relu) {
  const int lane = threadIdx.x & 31;
  const int wid  = blockIdx.x * (blockDim.x >> 5) + (threadIdx.x >> 5);
  const int nt64 = N >> 6;
  const int total = (Mp >> 4) * nt64;
  if (wid >= total) return;                    // wave-uniform: safe around WMMA
  const int mt = wid / nt64;
  const int nt = wid % nt64;

  const int kb_a = (lane >> 4) << 4;           // A: K-chunk base 0 or 16
  const int kb_b = (lane >> 4) << 4;           // B: K-chunk base 0 or 16
  const unsigned short* ap  = A + (size_t)(mt * 16 + (lane & 15)) * K + kb_a;
  const int wn = nt * 64 + (lane & 15);
  const unsigned short* wp0 = W + (size_t)(wn +  0) * K + kb_b;
  const unsigned short* wp1 = W + (size_t)(wn + 16) * K + kb_b;
  const unsigned short* wp2 = W + (size_t)(wn + 32) * K + kb_b;
  const unsigned short* wp3 = W + (size_t)(wn + 48) * K + kb_b;

  floatx8 acc0 = {}, acc1 = {}, acc2 = {}, acc3 = {};
  for (int k0 = 0; k0 < K; k0 += 32) {
    BFrag a, b0, b1, b2, b3;
    a.q[0]  = *(const u32x4*)(ap  + k0);  a.q[1]  = *(const u32x4*)(ap  + k0 + 8);
    b0.q[0] = *(const u32x4*)(wp0 + k0);  b0.q[1] = *(const u32x4*)(wp0 + k0 + 8);
    b1.q[0] = *(const u32x4*)(wp1 + k0);  b1.q[1] = *(const u32x4*)(wp1 + k0 + 8);
    b2.q[0] = *(const u32x4*)(wp2 + k0);  b2.q[1] = *(const u32x4*)(wp2 + k0 + 8);
    b3.q[0] = *(const u32x4*)(wp3 + k0);  b3.q[1] = *(const u32x4*)(wp3 + k0 + 8);
    acc0 = __builtin_amdgcn_wmma_f32_16x16x32_bf16(false, a.v, false, b0.v, (short)0, acc0, false, false);
    acc1 = __builtin_amdgcn_wmma_f32_16x16x32_bf16(false, a.v, false, b1.v, (short)0, acc1, false, false);
    acc2 = __builtin_amdgcn_wmma_f32_16x16x32_bf16(false, a.v, false, b2.v, (short)0, acc2, false, false);
    acc3 = __builtin_amdgcn_wmma_f32_16x16x32_bf16(false, a.v, false, b3.v, (short)0, acc3, false, false);
  }

  const int n_l = lane & 15;
  const int m_b = mt * 16 + ((lane >> 4) << 3);
  floatx8 accs[4] = {acc0, acc1, acc2, acc3};
#pragma unroll
  for (int j = 0; j < 4; ++j) {
    const int c = nt * 64 + j * 16 + n_l;
    const float bv = bias ? bias[c] : 0.0f;
#pragma unroll
    for (int r = 0; r < 8; ++r) {
      const int m = m_b + r;
      if (m < Mv) {
        float v = accs[j][r] + bv;
        if (addend)  v += addend[(size_t)m * N + c];
        if (do_relu) v = fmaxf(v, 0.0f);
        C[(size_t)m * N + c] = v;
      }
    }
  }
}

// ---------------- fused GRU recurrent step (one time step, all 256 mols) ----------------
// gh = h @ w_hh^T + b_hh (3 gate column-blocks via 3 WMMA accumulators), then gate math.
__global__ __launch_bounds__(256) void
k_gru_step(const unsigned short* __restrict__ h_bf, const float* __restrict__ h_f,
           const unsigned short* __restrict__ w_hh, const float* __restrict__ b_hh,
           const float* __restrict__ gi, int tx, int col_off,
           float* __restrict__ h_out_f, unsigned short* __restrict__ h_out_bf,
           unsigned short* __restrict__ gmsg) {
  const int lane = threadIdx.x & 31;
  const int wid  = blockIdx.x * (blockDim.x >> 5) + (threadIdx.x >> 5);   // 256 waves
  const int mt = wid >> 4, nt = wid & 15;
  const int K = H;
  const int kb_a = (lane >> 4) << 4;
  const int kb_b = (lane >> 4) << 4;
  const unsigned short* ap = h_bf + (size_t)(mt * 16 + (lane & 15)) * K + kb_a;
  const int wn = nt * 16 + (lane & 15);
  const unsigned short* wr = w_hh + (size_t)(wn)           * K + kb_b;
  const unsigned short* wz = w_hh + (size_t)(wn + H)       * K + kb_b;
  const unsigned short* wg = w_hh + (size_t)(wn + 2 * H)   * K + kb_b;

  floatx8 aR = {}, aZ = {}, aN = {};
  for (int k0 = 0; k0 < K; k0 += 32) {
    BFrag a, br, bz, bn;
    a.q[0]  = *(const u32x4*)(ap + k0);  a.q[1]  = *(const u32x4*)(ap + k0 + 8);
    br.q[0] = *(const u32x4*)(wr + k0);  br.q[1] = *(const u32x4*)(wr + k0 + 8);
    bz.q[0] = *(const u32x4*)(wz + k0);  bz.q[1] = *(const u32x4*)(wz + k0 + 8);
    bn.q[0] = *(const u32x4*)(wg + k0);  bn.q[1] = *(const u32x4*)(wg + k0 + 8);
    aR = __builtin_amdgcn_wmma_f32_16x16x32_bf16(false, a.v, false, br.v, (short)0, aR, false, false);
    aZ = __builtin_amdgcn_wmma_f32_16x16x32_bf16(false, a.v, false, bz.v, (short)0, aZ, false, false);
    aN = __builtin_amdgcn_wmma_f32_16x16x32_bf16(false, a.v, false, bn.v, (short)0, aN, false, false);
  }

  const int c   = nt * 16 + (lane & 15);
  const int m_b = mt * 16 + ((lane >> 4) << 3);
  const float br_ = b_hh[c], bz_ = b_hh[H + c], bn_ = b_hh[2 * H + c];
#pragma unroll
  for (int r = 0; r < 8; ++r) {
    const int mol = m_b + r;
    const float* gir = gi + (size_t)(mol * ASZ + tx) * (3 * H);
    const float ir = gir[c], iz = gir[H + c], in_ = gir[2 * H + c];
    const float rg = 1.0f / (1.0f + __expf(-(ir + aR[r] + br_)));
    const float zg = 1.0f / (1.0f + __expf(-(iz + aZ[r] + bz_)));
    const float ng = tanhf(in_ + rg * (aN[r] + bn_));
    const float hp = h_f[(size_t)mol * H + c];
    const float hn = (1.0f - zg) * ng + zg * hp;
    h_out_f[(size_t)mol * H + c]  = hn;
    h_out_bf[(size_t)mol * H + c] = f2bf(hn);
    gmsg[(size_t)(1 + mol * ASZ + tx) * (2 * H) + col_off + c] = f2bf(hn);
  }
}

// ---------------- host orchestration ----------------
static inline unsigned cdiv(long a, long b) { return (unsigned)((a + b - 1) / b); }

extern "C" void kernel_launch(void* const* d_in, const int* in_sizes, int n_in,
                              void* d_out, int out_size, void* d_ws, size_t ws_size,
                              hipStream_t stream) {
  (void)in_sizes; (void)n_in; (void)out_size; (void)ws_size;
  const float* f_atoms  = (const float*)d_in[0];
  const float* f_bonds  = (const float*)d_in[1];
  const int*   a2b      = (const int*)d_in[2];
  const int*   b2a      = (const int*)d_in[3];
  const int*   b2revb   = (const int*)d_in[4];
  const float* W_i_atom = (const float*)d_in[7];
  const float* W_i_bond = (const float*)d_in[8];
  const float* W_h_0    = (const float*)d_in[9];
  const float* W_h_1    = (const float*)d_in[10];
  const float* lr_W     = (const float*)d_in[11];
  const float* gru_bias = (const float*)d_in[12];
  const float* w_ih_f   = (const float*)d_in[13];
  const float* w_hh_f   = (const float*)d_in[14];
  const float* b_ih_f   = (const float*)d_in[15];
  const float* b_hh_f   = (const float*)d_in[16];
  const float* w_ih_b   = (const float*)d_in[17];
  const float* w_hh_b   = (const float*)d_in[18];
  const float* b_ih_b   = (const float*)d_in[19];
  const float* b_hh_b   = (const float*)d_in[20];
  const float* W_o_W    = (const float*)d_in[21];
  const float* W_o_b    = (const float*)d_in[22];
  const float* ln_g     = (const float*)d_in[23];
  const float* ln_b     = (const float*)d_in[24];
  float* out = (float*)d_out;
  char*  ws  = (char*)d_ws;

  constexpr size_t MiB = 1ull << 20;
  unsigned short* fb_bf  = (unsigned short*)(ws + 0 * MiB);    // 41 MiB region
  unsigned short* diffbf = (unsigned short*)(ws + 0 * MiB);    // reuse (fb dead after G2)
  float*          ib     = (float*)(ws + 41 * MiB);            // 65 MiB
  float*          mb     = (float*)(ws + 106 * MiB);           // 65 MiB
  unsigned short* fa_bf  = (unsigned short*)(ws + 171 * MiB);  // 9 MiB
  unsigned short* msg_bf = (unsigned short*)(ws + 180 * MiB);  // 8 MiB
  float*          ia     = (float*)(ws + 188 * MiB);           // 17 MiB
  float*          ma     = (float*)(ws + 205 * MiB);
  float*          aggm   = (float*)(ws + 222 * MiB);
  float*          node   = (float*)(ws + 239 * MiB);
  unsigned short* cat_bf = (unsigned short*)(ws + 256 * MiB);  // 25 MiB
  float*          msg    = (float*)(ws + 281 * MiB);
  float*          gi_f   = (float*)(ws + 298 * MiB);           // 48 MiB
  float*          gi_b   = (float*)(ws + 346 * MiB);           // 48 MiB
  unsigned short* gmsg   = (unsigned short*)(ws + 394 * MiB);  // 17 MiB
  float*          ah     = (float*)(ws + 411 * MiB);           // 17 MiB
  char* sm = ws + 428 * MiB;                                   // small region (~4 MiB)
  unsigned short* wia_bf  = (unsigned short*)sm; sm += 256 * 256 * 2;
  unsigned short* wib_bf  = (unsigned short*)sm; sm += 256 * 320 * 2;
  unsigned short* wh0_bf  = (unsigned short*)sm; sm += 256 * 256 * 2;
  unsigned short* wh1_bf  = (unsigned short*)sm; sm += 256 * 256 * 2;
  unsigned short* lrw_bf  = (unsigned short*)sm; sm += 256 * 768 * 2;
  unsigned short* wihf_bf = (unsigned short*)sm; sm += 768 * 256 * 2;
  unsigned short* whhf_bf = (unsigned short*)sm; sm += 768 * 256 * 2;
  unsigned short* wihb_bf = (unsigned short*)sm; sm += 768 * 256 * 2;
  unsigned short* whhb_bf = (unsigned short*)sm; sm += 768 * 256 * 2;
  unsigned short* wo_bf   = (unsigned short*)sm; sm += 256 * 512 * 2;
  float*          h0f     = (float*)sm;          sm += 256 * 256 * 4;
  unsigned short* h0bf    = (unsigned short*)sm; sm += 256 * 256 * 2;
  float*          hppf0   = (float*)sm;          sm += 256 * 256 * 4;
  float*          hppf1   = (float*)sm;          sm += 256 * 256 * 4;
  unsigned short* hppb0   = (unsigned short*)sm; sm += 256 * 256 * 2;
  unsigned short* hppb1   = (unsigned short*)sm; sm += 256 * 256 * 2;

  const int T = 256;
  auto gemm = [&](const unsigned short* A, const unsigned short* W, const float* bias,
                  const float* add, float* C, int Mp, int Mv, int N, int K, int relu) {
    long waves = (long)(Mp / 16) * (N / 64);
    k_gemm<<<cdiv(waves, 8), T, 0, stream>>>(A, W, bias, add, C, Mp, Mv, N, K, relu);
  };

  // --- bf16 conversions (activations padded, weights flat) ---
  k_conv_pad<<<cdiv((long)MP_A * H, T), T, 0, stream>>>(f_atoms, fa_bf, NA, MP_A, H);
  k_conv_pad<<<cdiv((long)MP_B * BOND_D, T), T, 0, stream>>>(f_bonds, fb_bf, NB, MP_B, BOND_D);
  k_conv_flat<<<cdiv(256 * 256, T), T, 0, stream>>>(W_i_atom, wia_bf, 256 * 256);
  k_conv_flat<<<cdiv(256 * 320, T), T, 0, stream>>>(W_i_bond, wib_bf, 256 * 320);
  k_conv_flat<<<cdiv(256 * 256, T), T, 0, stream>>>(W_h_0, wh0_bf, 256 * 256);
  k_conv_flat<<<cdiv(256 * 256, T), T, 0, stream>>>(W_h_1, wh1_bf, 256 * 256);
  k_conv_flat<<<cdiv(256 * 768, T), T, 0, stream>>>(lr_W, lrw_bf, 256 * 768);
  k_conv_flat<<<cdiv(768 * 256, T), T, 0, stream>>>(w_ih_f, wihf_bf, 768 * 256);
  k_conv_flat<<<cdiv(768 * 256, T), T, 0, stream>>>(w_hh_f, whhf_bf, 768 * 256);
  k_conv_flat<<<cdiv(768 * 256, T), T, 0, stream>>>(w_ih_b, wihb_bf, 768 * 256);
  k_conv_flat<<<cdiv(768 * 256, T), T, 0, stream>>>(w_hh_b, whhb_bf, 768 * 256);
  k_conv_flat<<<cdiv(256 * 512, T), T, 0, stream>>>(W_o_W, wo_bf, 256 * 512);

  // --- input projections ---
  gemm(fa_bf, wia_bf, nullptr, nullptr, ia, MP_A, NA, H, H, 1);        // input_atom
  gemm(fb_bf, wib_bf, nullptr, nullptr, ib, MP_B, NB, H, BOND_D, 1);   // input_bond

  // --- 2 message-passing steps ---
  const float* ma_in  = ia;
  const float* mb_src = ib;
  const unsigned short* wh[2] = {wh0_bf, wh1_bf};
  for (int d = 0; d < 2; ++d) {
    k_agg<<<cdiv((long)NA * H, T), T, 0, stream>>>(mb_src, a2b, ma_in, ma, NA);
    k_gatherdiff<<<cdiv((long)MP_B * H, T), T, 0, stream>>>(ma, mb_src, b2a, b2revb, diffbf);
    gemm(diffbf, wh[d], nullptr, ib, mb, MP_B, NB, H, H, 1);           // relu(ib + diff@Wh^T)
    ma_in = ma; mb_src = mb;
  }
  k_agg<<<cdiv((long)NA * H, T), T, 0, stream>>>(mb, a2b, nullptr, aggm, NA);

  // --- node = cat3 @ lr_W^T ---
  k_concat<<<cdiv((long)MP_A * 3 * H, T), T, 0, stream>>>(aggm, ma, ia, cat_bf);
  gemm(cat_bf, lrw_bf, nullptr, nullptr, node, MP_A, NA, H, 3 * H, 0);

  // --- BatchGRU ---
  k_msg<<<cdiv((long)NA * H, T), T, 0, stream>>>(node, gru_bias, msg, msg_bf);
  k_h0<<<cdiv(NMOL * H, T), T, 0, stream>>>(node, h0f, h0bf);
  gemm(msg_bf, wihf_bf, b_ih_f, nullptr, gi_f, NMOL * ASZ, NMOL * ASZ, 3 * H, H, 0);
  gemm(msg_bf, wihb_bf, b_ih_b, nullptr, gi_b, NMOL * ASZ, NMOL * ASZ, 3 * H, H, 0);

  {  // forward direction
    const float* hf = h0f; const unsigned short* hb = h0bf;
    for (int t = 0; t < ASZ; ++t) {
      float* hfo = (t & 1) ? hppf1 : hppf0;
      unsigned short* hbo = (t & 1) ? hppb1 : hppb0;
      k_gru_step<<<32, T, 0, stream>>>(hb, hf, whhf_bf, b_hh_f, gi_f, t, 0, hfo, hbo, gmsg);
      hf = hfo; hb = hbo;
    }
  }
  {  // backward direction
    const float* hf = h0f; const unsigned short* hb = h0bf;
    for (int t = 0; t < ASZ; ++t) {
      int tx = ASZ - 1 - t;
      float* hfo = (t & 1) ? hppf1 : hppf0;
      unsigned short* hbo = (t & 1) ? hppb1 : hppb0;
      k_gru_step<<<32, T, 0, stream>>>(hb, hf, whhb_bf, b_hh_b, gi_b, tx, H, hfo, hbo, gmsg);
      hf = hfo; hb = hbo;
    }
  }
  k_gmsg_fixup<<<cdiv((long)(1 + MP_A - NA) * 2 * H, T), T, 0, stream>>>(msg, gmsg);

  // --- output projection + residual + LayerNorm ---
  gemm(gmsg, wo_bf, W_o_b, nullptr, ah, MP_A, NA, H, 2 * H, 1);
  k_ln<<<NA, H, 0, stream>>>(f_atoms, ah, ln_g, ln_b, out);
}
